// Decoder_24438363914350
// MI455X (gfx1250) — matrix operand order repaired and checked by reading, not compiled, not run on hardware
//
#include <hip/hip_runtime.h>
#include <hip/hip_bf16.h>
#include <math.h>

typedef __attribute__((ext_vector_type(16))) _Float16 v16h;
typedef __attribute__((ext_vector_type(8)))  _Float16 v8h;
typedef __attribute__((ext_vector_type(8)))  float    v8f;

#define R_TOT   16384   // B*N = 32*512
#define NNODES  512
#define BATCH   32
#define TSTEPS  12
#define HID     64
#define FIN     2
#define MSUP    4

// ---------------------------------------------------------------------------
// diag[m,n] = adj[m,n,n]
// ---------------------------------------------------------------------------
__global__ void diag_extract(const float* __restrict__ adj, float* __restrict__ diag) {
    int idx = blockIdx.x * blockDim.x + threadIdx.x;         // m*512+n
    if (idx >= MSUP * NNODES) return;
    int m = idx >> 9, n = idx & 511;
    diag[idx] = adj[(m * NNODES + n) * NNODES + n];
}

// ---------------------------------------------------------------------------
// Pack fp32 weights [Korig x O] into f16 WMMA B-fragment layout, zero-padding
// K rows up to Kpad.  Fragment layout: lane holds column (lane&15); lanes<16
// hold K = kt*32+0..15 in elements 0..15, lanes>=16 hold K = kt*32+16..31.
// Bp[(((kt*(O/16)) + ot)*32 + lane)*16 + e]
// ---------------------------------------------------------------------------
__global__ void pack_weights(const float* __restrict__ W, _Float16* __restrict__ Bp,
                             int Korig, int O) {
    int K   = blockIdx.y;                 // 0..Kpad-1
    int col = blockIdx.x * blockDim.x + threadIdx.x;
    if (col >= O) return;
    float v = (K < Korig) ? W[K * O + col] : 0.f;
    int kt   = K >> 5;
    int klo  = K & 31;
    int lane = ((klo >> 4) << 4) | (col & 15);
    int e    = klo & 15;
    int ot   = col >> 4;
    Bp[(((kt * (O >> 4) + ot) << 5) + lane) * 16 + e] = (_Float16)v;
}

// ---------------------------------------------------------------------------
// Build expanded/diag-scaled input  dst[row, m*Kb+f] (f16), zero-padded to Kpad.
//   f <  Fx : "x" part  (xmode 0: targets slice tm1, or zeros if tm1<0;
//                        xmode 1: f32 buffer [R, Fx])
//   f >= Fx : h part, optionally scaled by r = gates[row*128 + (f-Fx)]
// ---------------------------------------------------------------------------
__global__ void build_xg(_Float16* __restrict__ dst, int Kpad, int Kb, int Fx,
                         const float* __restrict__ xsrc, int xmode, int tm1,
                         const float* __restrict__ hsrc,
                         const float* __restrict__ rgates,
                         const float* __restrict__ diag) {
    int col = blockIdx.x * blockDim.x + threadIdx.x;
    if (col >= Kpad) return;
    int row = blockIdx.y;
    float v = 0.f;
    if (col < MSUP * Kb) {
        int m = col / Kb;
        int f = col - m * Kb;
        if (f < Fx) {
            if (xmode == 0) {
                if (tm1 >= 0) {
                    int b = row >> 9, n = row & 511;
                    v = xsrc[((b * TSTEPS + tm1) * NNODES + n) * FIN + f];
                }
            } else {
                v = xsrc[row * Fx + f];
            }
        } else {
            float hv = hsrc[row * HID + (f - Fx)];
            if (rgates) hv *= rgates[row * 128 + (f - Fx)];
            v = hv;
        }
        v *= diag[m * NNODES + (row & 511)];
    }
    dst[row * Kpad + col] = (_Float16)v;
}

// ---------------------------------------------------------------------------
// WMMA GEMM: C[R x O] = A[R x K](f16) * Bp(packed f16) + bias, with epilogue:
//   epi==0 : sigmoid -> gates[row*128 + col]
//   epi==1 : cand = tanh(acc+bias); u = gates[row*128+64+col];
//            h[row*64+col] = u*h + (1-u)*cand        (GRU update in place)
// Block: 256 threads = 8 waves (4x2), wave tile 32x32 (2x2 WMMA), block 128x64.
// ---------------------------------------------------------------------------
__global__ __launch_bounds__(256)
void gemm_wmma(const _Float16* __restrict__ A, const _Float16* __restrict__ Bp,
               int K, int O, const float* __restrict__ bias,
               float* __restrict__ gates, float* __restrict__ hbuf, int epi) {
    const int lane = threadIdx.x & 31;
    const int w    = threadIdx.x >> 5;
    const int wr   = w >> 1;                  // 0..3
    const int wc   = w & 1;                   // 0..1
    const int rowBase = blockIdx.y * 128 + wr * 32;
    const int colBase = blockIdx.x * 64  + wc * 32;
    if (colBase >= O) return;

    v8f c00 = {}, c01 = {}, c10 = {}, c11 = {};

    const int arow0 = rowBase + (lane & 15);
    const int arow1 = arow0 + 16;
    const int koff  = (lane >> 4) << 4;       // 0 or 16
    const int nOT   = O >> 4;
    const int ot0   = colBase >> 4;

    const int kt_end = K >> 5;
    for (int kt = 0; kt < kt_end; ++kt) {
        const int k0 = kt << 5;
        union { v16h v; v8h h[2]; } a0, a1;
        const _Float16* pr0 = A + arow0 * K + k0 + koff;
        const _Float16* pr1 = A + arow1 * K + k0 + koff;
        a0.h[0] = *(const v8h*)(pr0);
        a0.h[1] = *(const v8h*)(pr0 + 8);
        a1.h[0] = *(const v8h*)(pr1);
        a1.h[1] = *(const v8h*)(pr1 + 8);

        const _Float16* bb = Bp + ((kt * nOT + ot0) * 32 + lane) * 16;
        v16h b0 = *(const v16h*)(bb);
        v16h b1 = *(const v16h*)(bb + 32 * 16);

        c00 = __builtin_amdgcn_wmma_f32_16x16x32_f16(false, a0.v, false, b0, (short)0, c00, false, false);
        c01 = __builtin_amdgcn_wmma_f32_16x16x32_f16(false, a0.v, false, b1, (short)0, c01, false, false);
        c10 = __builtin_amdgcn_wmma_f32_16x16x32_f16(false, a1.v, false, b0, (short)0, c10, false, false);
        c11 = __builtin_amdgcn_wmma_f32_16x16x32_f16(false, a1.v, false, b1, (short)0, c11, false, false);
    }

    // Epilogue: C/D layout -> row = base + i + 8*(lane>=16), col = base + (lane&15)
    const int rOff = (lane >> 4) << 3;
    const int cl   = lane & 15;
#pragma unroll
    for (int q = 0; q < 4; ++q) {
        v8f acc = (q == 0) ? c00 : (q == 1) ? c01 : (q == 2) ? c10 : c11;
        const int rt = (q >> 1) * 16;
        const int ct = (q & 1) * 16;
        const int col = colBase + ct + cl;
#pragma unroll
        for (int i = 0; i < 8; ++i) {
            const int row = rowBase + rt + rOff + i;
            float z = acc[i] + bias[col];
            if (epi == 0) {
                gates[row * 128 + col] = 1.f / (1.f + expf(-z));
            } else {
                float cand = tanhf(z);
                float u = gates[row * 128 + 64 + col];
                float h = hbuf[row * HID + col];
                hbuf[row * HID + col] = u * h + (1.f - u) * cand;
            }
        }
    }
}

// ---------------------------------------------------------------------------
// out[b,t,n,:] = h1[row,:] @ W_out[64x2] + b_out
// ---------------------------------------------------------------------------
__global__ void out_proj(const float* __restrict__ h1, const float* __restrict__ Wout,
                         const float* __restrict__ bout, float* __restrict__ out, int t) {
    int row = blockIdx.x * blockDim.x + threadIdx.x;
    if (row >= R_TOT) return;
    int b = row >> 9, n = row & 511;
    float a0 = bout[0], a1 = bout[1];
#pragma unroll 8
    for (int k = 0; k < HID; ++k) {
        float hv = h1[row * HID + k];
        a0 += hv * Wout[k * 2 + 0];
        a1 += hv * Wout[k * 2 + 1];
    }
    size_t base = ((size_t)(b * TSTEPS + t) * NNODES + n) * FIN;
    out[base + 0] = a0;
    out[base + 1] = a1;
}

// ---------------------------------------------------------------------------
extern "C" void kernel_launch(void* const* d_in, const int* in_sizes, int n_in,
                              void* d_out, int out_size, void* d_ws, size_t ws_size,
                              hipStream_t stream) {
    const float* h_init  = (const float*)d_in[0];
    const float* targets = (const float*)d_in[1];
    const float* adj     = (const float*)d_in[2];
    const float* W_g0    = (const float*)d_in[3];
    const float* b_g0    = (const float*)d_in[4];
    const float* W_c0    = (const float*)d_in[5];
    const float* b_c0    = (const float*)d_in[6];
    const float* W_g1    = (const float*)d_in[7];
    const float* b_g1    = (const float*)d_in[8];
    const float* W_c1    = (const float*)d_in[9];
    const float* b_c1    = (const float*)d_in[10];
    const float* W_out   = (const float*)d_in[11];
    const float* b_out   = (const float*)d_in[12];
    float* out = (float*)d_out;

    // workspace carve-up (all offsets 256B aligned)
    char* ws = (char*)d_ws;
    size_t off = 0;
    float*     diag  = (float*)(ws + off);     off += 4 * 512 * 4;               // 8 KB
    _Float16*  Wg0p  = (_Float16*)(ws + off);  off += (size_t)288 * 128 * 2;
    _Float16*  Wc0p  = (_Float16*)(ws + off);  off += (size_t)288 * 64  * 2;
    _Float16*  Wg1p  = (_Float16*)(ws + off);  off += (size_t)512 * 128 * 2;
    _Float16*  Wc1p  = (_Float16*)(ws + off);  off += (size_t)512 * 64  * 2;
    _Float16*  xg    = (_Float16*)(ws + off);  off += (size_t)R_TOT * 512 * 2;   // 16 MB
    float*     gates = (float*)(ws + off);     off += (size_t)R_TOT * 128 * 4;   // 8 MB
    float*     h0    = (float*)(ws + off);     off += (size_t)R_TOT * HID * 4;   // 4 MB
    float*     h1    = (float*)(ws + off);     off += (size_t)R_TOT * HID * 4;   // 4 MB
    (void)ws_size; (void)in_sizes; (void)n_in; (void)out_size;

    // ---- one-time setup ----
    diag_extract<<<(MSUP * NNODES + 255) / 256, 256, 0, stream>>>(adj, diag);
    pack_weights<<<dim3(1, 288), 128, 0, stream>>>(W_g0, Wg0p, 264, 128);
    pack_weights<<<dim3(1, 288), 128, 0, stream>>>(W_c0, Wc0p, 264, 64);
    pack_weights<<<dim3(1, 512), 128, 0, stream>>>(W_g1, Wg1p, 512, 128);
    pack_weights<<<dim3(1, 512), 128, 0, stream>>>(W_c1, Wc1p, 512, 64);
    hipMemcpyAsync(h0, h_init, (size_t)R_TOT * HID * 4, hipMemcpyDeviceToDevice, stream);
    hipMemcpyAsync(h1, h_init, (size_t)R_TOT * HID * 4, hipMemcpyDeviceToDevice, stream);

    const dim3 blk(256);
    const dim3 gB0(2, R_TOT);          // build for Kpad=288 (2 col-blocks of 256)
    const dim3 gB1(2, R_TOT);          // build for Kpad=512
    const dim3 gG128(2, R_TOT / 128);  // GEMM O=128
    const dim3 gG64(1, R_TOT / 128);   // GEMM O=64

    for (int t = 0; t < TSTEPS; ++t) {
        const int tm1 = t - 1;
        // ---- layer 0 ----
        build_xg<<<gB0, blk, 0, stream>>>(xg, 288, 66, FIN, targets, 0, tm1, h0, nullptr, diag);
        gemm_wmma<<<gG128, blk, 0, stream>>>(xg, Wg0p, 288, 128, b_g0, gates, nullptr, 0);
        build_xg<<<gB0, blk, 0, stream>>>(xg, 288, 66, FIN, targets, 0, tm1, h0, gates, diag);
        gemm_wmma<<<gG64, blk, 0, stream>>>(xg, Wc0p, 288, 64, b_c0, gates, h0, 1);
        // ---- layer 1 ----
        build_xg<<<gB1, blk, 0, stream>>>(xg, 512, 128, HID, h0, 1, 0, h1, nullptr, diag);
        gemm_wmma<<<gG128, blk, 0, stream>>>(xg, Wg1p, 512, 128, b_g1, gates, nullptr, 0);
        build_xg<<<gB1, blk, 0, stream>>>(xg, 512, 128, HID, h0, 1, 0, h1, gates, diag);
        gemm_wmma<<<gG64, blk, 0, stream>>>(xg, Wc1p, 512, 64, b_c1, gates, h1, 1);
        // ---- projection ----
        out_proj<<<(R_TOT + 255) / 256, 256, 0, stream>>>(h1, W_out, b_out, out, t);
    }
}
